// PoseRefineHeadSparseConv_31696858644901
// MI455X (gfx1250) — compile-verified, matrix operand order checked by reading, not run on hardware
//
#include <hip/hip_runtime.h>
#include <climits>

// ---------------------------------------------------------------------------
// PoseRefineHeadSparseConv on MI455X (gfx1250, wave32, WMMA + TDM)
//  - all GEMM-like stages on v_wmma_f32_16x16x32_f16 (f16 in, f32 acc)
//  - subm-conv neighbor gather done by the Tensor Data Mover in gather mode,
//    double-buffered into LDS, shared by the 12 N-tile waves of a workgroup
// ---------------------------------------------------------------------------

#define NPTS 4096
#define CF   192
#define NH   12
#define DH   16
#define NLAY 2
#define GRD  40
#define G3V  (GRD*GRD*GRD)
#define VOXSZ 0.1f
#define CB_K (CF/32)     // 6 K-blocks of 32
#define NT_N (CF/16)     // 12 N-tiles of 16
#define VT_M (NPTS/16)   // 256 M-tiles of 16
#define FRAG_ELEMS 512   // 32 lanes * 16 halves per B fragment

typedef __attribute__((ext_vector_type(16))) _Float16 v16h;
typedef __attribute__((ext_vector_type(8)))  _Float16 v8h;
typedef __attribute__((ext_vector_type(8)))  float    v8f;
typedef __attribute__((ext_vector_type(4)))  unsigned int u32x4;
typedef __attribute__((ext_vector_type(8)))  int          i32x8;
typedef __attribute__((ext_vector_type(4)))  int          i32x4;

// Per-lane element i of a 16-bit WMMA A/B fragment maps to K index:
__device__ __forceinline__ int frag_k(int i, int half) {
  return (i < 8) ? (half * 8 + i) : (16 + half * 8 + (i - 8));
}

// -------------------- TDM gather: 16 voxel rows (192 f16 each) -> LDS --------------------
// D# gather mode: 16-bit row indices; index 0xFFFF (from nidx == -1) is OOB
// (>= tensor_dim1) and therefore returns zeros == submanifold "inactive" rows.
__device__ __forceinline__ void tdm_gather_rows(const _Float16* xh, const int* nidx,
                                                int vt, int k, const _Float16* ldsdst) {
  unsigned lds_off = (unsigned)(size_t)ldsdst;
  unsigned long long ga = (unsigned long long)(size_t)xh;
  unsigned short idx[16];
#pragma unroll
  for (int r = 0; r < 16; r++)
    idx[r] = (unsigned short)nidx[(vt * 16 + r) * 27 + k];   // -1 -> 0xFFFF -> OOB -> zeros

  u32x4 g0;
  g0[0] = 0x80000001u;                                  // count=1, gather_mode=1 (16-bit idx)
  g0[1] = lds_off;                                      // lds_addr
  g0[2] = (unsigned)(ga & 0xFFFFFFFFu);                 // global_addr[31:0]
  g0[3] = (unsigned)((ga >> 32) & 0x01FFFFFFu) | 0x80000000u;   // addr[56:32] | type=2

  i32x8 g1;
  g1[0] = 0x10000;                 // workgroup_mask=0, data_size=1 (2 bytes)
  g1[1] = (CF & 0xFFFF) << 16;     // tensor_dim0 = 192 (low 16 in bits 63:48)
  g1[2] = (NPTS & 0xFFFF) << 16;   // tensor_dim0 hi=0 | tensor_dim1 = 4096 (rows)
  g1[3] = CF << 16;                // tensor_dim1 hi=0 | tile_dim0 = 192
  g1[4] = 16;                      // tile_dim1 = 16 valid gather indices
  g1[5] = CF;                      // tensor_dim0_stride = 192 elements
  g1[6] = 0;
  g1[7] = 0;

  i32x4 g2, g3;
#pragma unroll
  for (int w = 0; w < 4; w++) {
    g2[w] = (int)((unsigned)idx[2 * w]     | ((unsigned)idx[2 * w + 1] << 16));
    g3[w] = (int)((unsigned)idx[8 + 2 * w] | ((unsigned)idx[8 + 2 * w + 1] << 16));
  }
#if __clang_major__ >= 23
  i32x8 g4;
#pragma unroll
  for (int w = 0; w < 8; w++) g4[w] = 0;
  __builtin_amdgcn_tensor_load_to_lds(g0, g1, g2, g3, g4, 0);
#else
  __builtin_amdgcn_tensor_load_to_lds(g0, g1, g2, g3, 0);
#endif
}

// -------------------- small utility kernels --------------------

__global__ void k_fill_i32(int* p, int n, int v) {
  int i = blockIdx.x * blockDim.x + threadIdx.x;
  if (i < n) p[i] = v;
}

__device__ float reduce192(float x, float* sh) {
  int c = threadIdx.x;
  sh[c] = x; __syncthreads();
  for (int s = 96; s > 1; s >>= 1) {
    if (c < s) sh[c] += sh[c + s];
    __syncthreads();
  }
  float tot = sh[0] + sh[1] + sh[2];
  __syncthreads();
  return tot;
}

// -------------------- encoder: relu(LN(points @ enc_w + enc_b)) --------------------

__global__ void k_encode(const float* __restrict__ pts, const float* __restrict__ ew,
                         const float* __restrict__ eb, const float* __restrict__ lg,
                         const float* __restrict__ lb, float* __restrict__ feats) {
  __shared__ float sh[CF];
  int p = blockIdx.x, c = threadIdx.x;
  float x = pts[p*3+0]*ew[c] + pts[p*3+1]*ew[CF+c] + pts[p*3+2]*ew[2*CF+c] + eb[c];
  float mu = reduce192(x, sh) * (1.0f / CF);
  float d  = x - mu;
  float var = reduce192(d * d, sh) * (1.0f / CF);
  float y = d * rsqrtf(var + 1e-5f) * lg[c] + lb[c];
  feats[(size_t)p * CF + c] = fmaxf(y, 0.f);
}

// -------------------- voxelization (deterministic) --------------------

__global__ void k_coordmin(const float* __restrict__ pts, int* cmin) {
  int i = blockIdx.x * blockDim.x + threadIdx.x;
  if (i >= NPTS) return;
  for (int a = 0; a < 3; a++) {
    int cc = (int)floorf(pts[i*3+a] / VOXSZ);
    atomicMin(&cmin[a], cc);
  }
}

__global__ void k_hashmark(const float* __restrict__ pts, const int* __restrict__ cmin,
                           int* __restrict__ hpt, int* __restrict__ lut) {
  int i = blockIdx.x * blockDim.x + threadIdx.x;
  if (i >= NPTS) return;
  int cx = (int)floorf(pts[i*3+0] / VOXSZ) - cmin[0];
  int cy = (int)floorf(pts[i*3+1] / VOXSZ) - cmin[1];
  int cz = (int)floorf(pts[i*3+2] / VOXSZ) - cmin[2];
  int h = (cx * GRD + cy) * GRD + cz;
  hpt[i] = h;
  lut[h] = 1;   // occupancy mark (benign race)
}

// Single-workgroup scan over G3V cells -> sorted unique voxel ids (jnp.unique order)
__global__ void k_scan(int* __restrict__ lut, int* __restrict__ vcoords, int* __restrict__ nvox) {
  __shared__ int sh[1024];
  __shared__ int carry;
  if (threadIdx.x == 0) carry = 0;
  __syncthreads();
  for (int base = 0; base < G3V; base += 1024) {
    int g = base + threadIdx.x;
    int occ = (g < G3V) ? (lut[g] != 0 ? 1 : 0) : 0;
    sh[threadIdx.x] = occ; __syncthreads();
    for (int off = 1; off < 1024; off <<= 1) {
      int v = (threadIdx.x >= (unsigned)off) ? sh[threadIdx.x - off] : 0;
      __syncthreads();
      sh[threadIdx.x] += v;
      __syncthreads();
    }
    int incl = sh[threadIdx.x];
    int id = carry + incl - occ;
    if (g < G3V) {
      if (occ && id < NPTS) {
        lut[g] = id;
        vcoords[id*3+0] = g / (GRD*GRD);
        vcoords[id*3+1] = (g / GRD) % GRD;
        vcoords[id*3+2] = g % GRD;
      } else {
        lut[g] = -1;
      }
    }
    __syncthreads();
    if (threadIdx.x == 1023) carry += incl;
    __syncthreads();
  }
  if (threadIdx.x == 0) { nvox[0] = carry; lut[G3V] = -1; }
}

__global__ void k_inv(const int* __restrict__ hpt, const int* __restrict__ lut,
                      int* __restrict__ inv, int* __restrict__ cnts) {
  int p = blockIdx.x * blockDim.x + threadIdx.x;
  if (p >= NPTS) return;
  int v = lut[hpt[p]];
  inv[p] = v;
  atomicAdd(&cnts[v], 1);   // integer atomic: deterministic value
}

// deterministic per-voxel mean pooling (fixed summation order over point index)
__global__ void k_voxmean(const float* __restrict__ feats, const int* __restrict__ inv,
                          const int* __restrict__ cnts, float* __restrict__ xv,
                          int* __restrict__ valid, _Float16* __restrict__ xvh) {
  __shared__ int sinv[NPTS];
  int v = blockIdx.x, c = threadIdx.x;
  for (int p = c; p < NPTS; p += blockDim.x) sinv[p] = inv[p];
  __syncthreads();
  int cnt = cnts[v];
  if (c == 0) valid[v] = (cnt > 0) ? 1 : 0;
  float s = 0.f;
  if (cnt > 0) {
    for (int p = 0; p < NPTS; p++)
      if (sinv[p] == v) s += feats[(size_t)p * CF + c];
    s /= (float)cnt;
  }
  xv[(size_t)v * CF + c]  = s;
  xvh[(size_t)v * CF + c] = (_Float16)s;
}

__global__ void k_neighbors(const int* __restrict__ vcoords, const int* __restrict__ valid,
                            const int* __restrict__ lut, int* __restrict__ nidx) {
  int v = blockIdx.x, t = threadIdx.x;
  if (t >= 27) return;
  int nval = -1;
  if (valid[v]) {
    int dx = t / 9 - 1, dy = (t / 3) % 3 - 1, dz = t % 3 - 1;
    int x = vcoords[v*3+0] + dx, y = vcoords[v*3+1] + dy, z = vcoords[v*3+2] + dz;
    if (x >= 0 && x < GRD && y >= 0 && y < GRD && z >= 0 && z < GRD)
      nval = lut[(x * GRD + y) * GRD + z];
  }
  nidx[v*27 + t] = nval;
}

// -------------------- weight pre-swizzle into B-fragment layout --------------------
__global__ void k_pack_frag(const float* __restrict__ w, _Float16* __restrict__ pack) {
  int f = blockIdx.x, lane = threadIdx.x;
  int nt = f % NT_N;
  int cb = (f / NT_N) % CB_K;
  int k  = f / (NT_N * CB_K);
  int n = lane & 15, half = lane >> 4;
  const float* wk = w + (size_t)k * CF * CF;
  _Float16* dst = pack + (size_t)f * FRAG_ELEMS + lane * 16;
  for (int i = 0; i < 16; i++) {
    int K = frag_k(i, half);
    dst[i] = (_Float16)wk[(size_t)(cb * 32 + K) * CF + nt * 16 + n];
  }
}

// -------------------- submanifold conv: TDM-gathered A, 12 waves share LDS --------------------

__global__ __launch_bounds__(384)
void k_subm_conv(const _Float16* __restrict__ xh, const int* __restrict__ nidx,
                 const _Float16* __restrict__ wpack, float* __restrict__ out) {
  __shared__ _Float16 Asm[2][16 * CF];     // double-buffered gathered neighbor rows (2 x 6 KB)
  int vt = blockIdx.x;
  int wid = threadIdx.x >> 5, lane = threadIdx.x & 31;
  int nt = wid;                            // each wave owns one 16-wide N tile
  int m = lane & 15, half = lane >> 4;
  v8f acc = {};

  if (threadIdx.x < 32) tdm_gather_rows(xh, nidx, vt, 0, &Asm[0][0]);

  for (int k = 0; k < 27; k++) {
    if (threadIdx.x < 32) {
      if (k + 1 < 27) {                    // overlap DMA of tap k+1 with compute of tap k
        tdm_gather_rows(xh, nidx, vt, k + 1, &Asm[(k + 1) & 1][0]);
        __builtin_amdgcn_s_wait_tensorcnt(1);
      } else {
        __builtin_amdgcn_s_wait_tensorcnt(0);
      }
    }
    __syncthreads();
    const _Float16* As = &Asm[k & 1][0];
    const _Float16* wk = wpack + (size_t)k * CB_K * NT_N * FRAG_ELEMS;
#pragma unroll
    for (int cb = 0; cb < CB_K; cb++) {
      v16h a;
      v8h lo = *(const v8h*)(As + m * CF + cb * 32 + half * 8);
      v8h hi = *(const v8h*)(As + m * CF + cb * 32 + 16 + half * 8);
#pragma unroll
      for (int i = 0; i < 8; i++) { a[i] = lo[i]; a[i + 8] = hi[i]; }
      v16h b = *(const v16h*)(wk + ((size_t)cb * NT_N + nt) * FRAG_ELEMS + lane * 16);
      acc = __builtin_amdgcn_wmma_f32_16x16x32_f16(false, a, false, b, (short)0, acc, false, false);
    }
    __syncthreads();                       // all waves done with Asm[k&1] before it is refilled
  }
#pragma unroll
  for (int r = 0; r < 8; r++)
    out[(size_t)(vt * 16 + r + 8 * half) * CF + nt * 16 + m] = acc[r];
}

// -------------------- masked BatchNorm --------------------

__global__ void k_bnreduce(const float* __restrict__ x, const int* __restrict__ valid,
                           float* __restrict__ bsum, float* __restrict__ bsq) {
  __shared__ float s1[256], s2[256];
  int c = blockIdx.x, t = threadIdx.x;
  float a = 0.f, b = 0.f;
  for (int v = t; v < NPTS; v += 256) {
    if (valid[v]) { float y = x[(size_t)v * CF + c]; a += y; b += y * y; }
  }
  s1[t] = a; s2[t] = b; __syncthreads();
  for (int s = 128; s >= 1; s >>= 1) {
    if (t < s) { s1[t] += s1[t + s]; s2[t] += s2[t + s]; }
    __syncthreads();
  }
  if (t == 0) { bsum[c] = s1[0]; bsq[c] = s2[0]; }
}

__global__ void k_bnapply(const float* __restrict__ x, const float* __restrict__ bsum,
                          const float* __restrict__ bsq, const int* __restrict__ nvox,
                          const float* __restrict__ g, const float* __restrict__ b,
                          const float* __restrict__ res, float* outf, _Float16* outh,
                          int do_relu) {
  int v = blockIdx.x, c = threadIdx.x;
  float n = fmaxf((float)nvox[0], 1.f);
  float mu = bsum[c] / n;
  float var = bsq[c] / n - mu * mu;
  size_t idx = (size_t)v * CF + c;
  float y = (x[idx] - mu) * rsqrtf(fmaxf(var, 0.f) + 1e-5f) * g[c] + b[c];
  if (res) y += res[idx];
  if (do_relu) y = fmaxf(y, 0.f);
  if (outf) outf[idx] = y;
  if (outh) outh[idx] = (_Float16)y;
}

__global__ void k_gatherpts(const float* __restrict__ xv, const int* __restrict__ inv,
                            float* __restrict__ pf, _Float16* __restrict__ pfh) {
  int p = blockIdx.x, c = threadIdx.x;
  float y = xv[(size_t)inv[p] * CF + c];
  pf[(size_t)p * CF + c]  = y;
  pfh[(size_t)p * CF + c] = (_Float16)y;
}

// -------------------- projection GEMM (X[4096x192] @ W[192x192] + b (+res)) --------------------
// outhT (optional) stores the f16 result transposed [CF x NPTS] (used for V).

__global__ __launch_bounds__(32)
void k_gemm_proj(const _Float16* __restrict__ xh, const _Float16* __restrict__ wpack,
                 const float* __restrict__ bias, const float* __restrict__ res,
                 float* outf, _Float16* outh, _Float16* outhT) {
  int rt = blockIdx.x, nt = blockIdx.y;
  int lane = threadIdx.x, m = lane & 15, half = lane >> 4;
  const _Float16* src = xh + (size_t)(rt * 16 + m) * CF;
  v8f acc = {};
  for (int cb = 0; cb < CB_K; cb++) {
    v16h a;
    v8h lo = *(const v8h*)(src + cb * 32 + half * 8);
    v8h hi = *(const v8h*)(src + cb * 32 + 16 + half * 8);
#pragma unroll
    for (int i = 0; i < 8; i++) { a[i] = lo[i]; a[i + 8] = hi[i]; }
    v16h b = *(const v16h*)(wpack + ((size_t)cb * NT_N + nt) * FRAG_ELEMS + lane * 16);
    acc = __builtin_amdgcn_wmma_f32_16x16x32_f16(false, a, false, b, (short)0, acc, false, false);
  }
  int ncol = nt * 16 + (lane & 15);
  float bv = bias ? bias[ncol] : 0.f;
#pragma unroll
  for (int r = 0; r < 8; r++) {
    int row = rt * 16 + r + 8 * half;
    size_t idx = (size_t)row * CF + ncol;
    float y = acc[r] + bv;
    if (res)   y += res[idx];
    if (outf)  outf[idx] = y;
    if (outh)  outh[idx] = (_Float16)y;
    if (outhT) outhT[(size_t)ncol * NPTS + row] = (_Float16)y;
  }
}

// -------------------- cross-attention: pass 1 (row max / sum-exp) --------------------

__global__ __launch_bounds__(32)
void k_attn_stats(const _Float16* __restrict__ qh, const _Float16* __restrict__ kh,
                  float* __restrict__ rowmax, float* __restrict__ rowsum) {
  int qt = blockIdx.x, h = blockIdx.y;
  int lane = threadIdx.x, m = lane & 15, half = lane >> 4;
  int hb = h * DH;
  v16h a;
  {
    v8h q8 = *(const v8h*)(qh + (size_t)(qt * 16 + m) * CF + hb + half * 8);
#pragma unroll
    for (int i = 0; i < 8; i++) { a[i] = q8[i]; a[i + 8] = (_Float16)0.f; }  // K padded 16->32
  }
  float lm[8], ls[8];
#pragma unroll
  for (int r = 0; r < 8; r++) { lm[r] = -3.0e38f; ls[r] = 0.f; }
  for (int mt = 0; mt < VT_M; mt++) {
    v16h b;
    v8h k8 = *(const v8h*)(kh + (size_t)(mt * 16 + m) * CF + hb + half * 8);
#pragma unroll
    for (int i = 0; i < 8; i++) { b[i] = k8[i]; b[i + 8] = (_Float16)0.f; }
    v8f z = {};
    v8f s = __builtin_amdgcn_wmma_f32_16x16x32_f16(false, a, false, b, (short)0, z, false, false);
#pragma unroll
    for (int r = 0; r < 8; r++) {
      float v = s[r] * 0.25f;                   // 1/sqrt(Dh)
      float nm = fmaxf(lm[r], v);
      ls[r] = ls[r] * __expf(lm[r] - nm) + __expf(v - nm);
      lm[r] = nm;
    }
  }
  for (int off = 1; off < 16; off <<= 1) {      // merge 16 key columns per half
#pragma unroll
    for (int r = 0; r < 8; r++) {
      float om = __shfl_xor(lm[r], off, 32);
      float os = __shfl_xor(ls[r], off, 32);
      float nm = fmaxf(lm[r], om);
      ls[r] = ls[r] * __expf(lm[r] - nm) + os * __expf(om - nm);
      lm[r] = nm;
    }
  }
  if ((lane & 15) == 0) {
#pragma unroll
    for (int r = 0; r < 8; r++) {
      int row = qt * 16 + r + 8 * half;
      rowmax[(size_t)row * NH + h] = lm[r];
      rowsum[(size_t)row * NH + h] = ls[r];
    }
  }
}

// -------------------- cross-attention: pass 2 (O = softmax(S) @ V) --------------------
// vT is the transposed V [CF x NPTS] so the B fragment is two contiguous 16B loads.

__global__ __launch_bounds__(32)
void k_attn_out(const _Float16* __restrict__ qh, const _Float16* __restrict__ kh,
                const _Float16* __restrict__ vT, const float* __restrict__ rowmax,
                const float* __restrict__ rowsum, _Float16* __restrict__ oh) {
  __shared__ _Float16 P[16 * 32];
  int qt = blockIdx.x, h = blockIdx.y;
  int lane = threadIdx.x, m = lane & 15, half = lane >> 4;
  int hb = h * DH;
  v16h a;
  {
    v8h q8 = *(const v8h*)(qh + (size_t)(qt * 16 + m) * CF + hb + half * 8);
#pragma unroll
    for (int i = 0; i < 8; i++) { a[i] = q8[i]; a[i + 8] = (_Float16)0.f; }
  }
  float rm[8], rs[8];
#pragma unroll
  for (int r = 0; r < 8; r++) {
    int row = qt * 16 + r + 8 * half;
    rm[r] = rowmax[(size_t)row * NH + h];
    rs[r] = rowsum[(size_t)row * NH + h];
  }
  v8f o = {};
  for (int mt0 = 0; mt0 < VT_M / 2; mt0++) {
#pragma unroll
    for (int sub = 0; sub < 2; sub++) {
      int mt = mt0 * 2 + sub;
      v16h b;
      v8h k8 = *(const v8h*)(kh + (size_t)(mt * 16 + m) * CF + hb + half * 8);
#pragma unroll
      for (int i = 0; i < 8; i++) { b[i] = k8[i]; b[i + 8] = (_Float16)0.f; }
      v8f z = {};
      v8f s = __builtin_amdgcn_wmma_f32_16x16x32_f16(false, a, false, b, (short)0, z, false, false);
#pragma unroll
      for (int r = 0; r < 8; r++) {
        float p = __expf(s[r] * 0.25f - rm[r]);
        P[(r + 8 * half) * 32 + sub * 16 + m] = (_Float16)p;  // D-layout: row=r+8*half, col=lane&15
      }
    }
    __syncthreads();
    v16h a2;                 // re-read P in A-layout (M=query rows, K=32 keys)
#pragma unroll
    for (int i = 0; i < 16; i++) a2[i] = P[m * 32 + frag_k(i, half)];
    v16h b2;                 // V fragment from vT: contiguous keys per lane
    {
      const _Float16* vrow = vT + (size_t)(hb + m) * NPTS + mt0 * 32;
      v8h x0 = *(const v8h*)(vrow + half * 8);
      v8h x1 = *(const v8h*)(vrow + 16 + half * 8);
#pragma unroll
      for (int i = 0; i < 8; i++) { b2[i] = x0[i]; b2[i + 8] = x1[i]; }
    }
    o = __builtin_amdgcn_wmma_f32_16x16x32_f16(false, a2, false, b2, (short)0, o, false, false);
    __syncthreads();
  }
#pragma unroll
  for (int r = 0; r < 8; r++)
    oh[(size_t)(qt * 16 + r + 8 * half) * CF + hb + m] = (_Float16)(o[r] / rs[r]);
}

// -------------------- LayerNorm / pooling / MLP --------------------

__global__ void k_layernorm(float* __restrict__ x, const float* __restrict__ g,
                            const float* __restrict__ b) {
  __shared__ float sh[CF];
  int p = blockIdx.x, c = threadIdx.x;
  size_t idx = (size_t)p * CF + c;
  float v = x[idx];
  float mu = reduce192(v, sh) * (1.0f / CF);
  float d = v - mu;
  float var = reduce192(d * d, sh) * (1.0f / CF);
  x[idx] = d * rsqrtf(var + 1e-5f) * g[c] + b[c];
}

__global__ void k_pool(const float* __restrict__ x, float* __restrict__ fused, int base) {
  __shared__ float smx[256], ssm[256];
  int c = blockIdx.x, t = threadIdx.x;
  float mx = -3.0e38f, sm = 0.f;
  for (int p = t; p < NPTS; p += 256) {
    float y = x[(size_t)p * CF + c];
    mx = fmaxf(mx, y); sm += y;
  }
  smx[t] = mx; ssm[t] = sm; __syncthreads();
  for (int s = 128; s >= 1; s >>= 1) {
    if (t < s) { smx[t] = fmaxf(smx[t], smx[t + s]); ssm[t] += ssm[t + s]; }
    __syncthreads();
  }
  if (t == 0) { fused[base + c] = smx[0]; fused[base + CF + c] = ssm[0] / (float)NPTS; }
}

__global__ void k_mlp(const float* __restrict__ in, const float* __restrict__ w,
                      const float* __restrict__ b, float* __restrict__ out,
                      int nin, int nout, int relu) {
  int j = blockIdx.x * blockDim.x + threadIdx.x;
  if (j >= nout) return;
  float s = b[j];
  for (int i = 0; i < nin; i++) s += in[i] * w[(size_t)i * nout + j];
  out[j] = relu ? fmaxf(s, 0.f) : s;
}

// ---------------------------------------------------------------------------

extern "C" void kernel_launch(void* const* d_in, const int* in_sizes, int n_in,
                              void* d_out, int out_size, void* d_ws, size_t ws_size,
                              hipStream_t stream) {
  (void)in_sizes; (void)n_in; (void)out_size; (void)ws_size;
  const float* src_pts  = (const float*)d_in[0];
  const float* tgt_pts  = (const float*)d_in[1];
  const float* enc_w    = (const float*)d_in[2];
  const float* enc_b    = (const float*)d_in[3];
  const float* enc_ln_g = (const float*)d_in[4];
  const float* enc_ln_b = (const float*)d_in[5];
  const float* conv1_w  = (const float*)d_in[6];
  const float* bn1_g    = (const float*)d_in[7];
  const float* bn1_b    = (const float*)d_in[8];
  const float* conv2_w  = (const float*)d_in[9];
  const float* bn2_g    = (const float*)d_in[10];
  const float* bn2_b    = (const float*)d_in[11];
  const float* q_w      = (const float*)d_in[12];
  const float* q_b      = (const float*)d_in[13];
  const float* k_w      = (const float*)d_in[14];
  const float* k_b      = (const float*)d_in[15];
  const float* v_w      = (const float*)d_in[16];
  const float* v_b      = (const float*)d_in[17];
  const float* o_w      = (const float*)d_in[18];
  const float* o_b      = (const float*)d_in[19];
  const float* aln_g    = (const float*)d_in[20];
  const float* aln_b    = (const float*)d_in[21];
  const float* h1_w     = (const float*)d_in[22];
  const float* h1_b     = (const float*)d_in[23];
  const float* h2_w     = (const float*)d_in[24];
  const float* h2_b     = (const float*)d_in[25];
  const float* h3_w     = (const float*)d_in[26];
  const float* h3_b     = (const float*)d_in[27];

  char* ws = (char*)d_ws;
  size_t cur = 0;
  auto alloc = [&](size_t bytes) -> void* {
    void* p = ws + cur;
    cur = (cur + bytes + 255) & ~(size_t)255;
    return p;
  };

  int*       lut     = (int*)alloc((G3V + 1) * sizeof(int));
  int*       hpt     = (int*)alloc(NPTS * sizeof(int));
  int*       inv     = (int*)alloc(NPTS * sizeof(int));
  int*       cnts    = (int*)alloc(NPTS * sizeof(int));
  int*       vcoords = (int*)alloc(3 * NPTS * sizeof(int));
  int*       valid   = (int*)alloc(NPTS * sizeof(int));
  int*       nidx    = (int*)alloc(27 * NPTS * sizeof(int));
  int*       cmin    = (int*)alloc(8 * sizeof(int));
  int*       nvox    = (int*)alloc(8 * sizeof(int));
  float*     feats   = (float*)alloc((size_t)NPTS * CF * sizeof(float));
  float*     xv      = (float*)alloc((size_t)NPTS * CF * sizeof(float));
  float*     tbuf    = (float*)alloc((size_t)NPTS * CF * sizeof(float));
  float*     pf0     = (float*)alloc((size_t)NPTS * CF * sizeof(float));
  float*     pf1     = (float*)alloc((size_t)NPTS * CF * sizeof(float));
  float*     xo      = (float*)alloc((size_t)NPTS * CF * sizeof(float));
  float*     rowmax  = (float*)alloc((size_t)NPTS * NH * sizeof(float));
  float*     rowsum  = (float*)alloc((size_t)NPTS * NH * sizeof(float));
  float*     bsum    = (float*)alloc(CF * sizeof(float));
  float*     bsq     = (float*)alloc(CF * sizeof(float));
  float*     fused   = (float*)alloc(4 * CF * sizeof(float));
  float*     hb1     = (float*)alloc(2 * CF * sizeof(float));
  float*     hb2     = (float*)alloc(CF * sizeof(float));
  _Float16*  xvh     = (_Float16*)alloc((size_t)NPTS * CF * sizeof(_Float16));
  _Float16*  o1h     = (_Float16*)alloc((size_t)NPTS * CF * sizeof(_Float16));
  _Float16*  pf0h    = (_Float16*)alloc((size_t)NPTS * CF * sizeof(_Float16));
  _Float16*  pf1h    = (_Float16*)alloc((size_t)NPTS * CF * sizeof(_Float16));
  _Float16*  qhb     = (_Float16*)alloc((size_t)NPTS * CF * sizeof(_Float16));
  _Float16*  khb     = (_Float16*)alloc((size_t)NPTS * CF * sizeof(_Float16));
  _Float16*  vT      = (_Float16*)alloc((size_t)NPTS * CF * sizeof(_Float16));
  _Float16*  ohb     = (_Float16*)alloc((size_t)NPTS * CF * sizeof(_Float16));
  const size_t convPackPerLayer = (size_t)27 * CB_K * NT_N * FRAG_ELEMS;  // halves
  _Float16*  cpack1  = (_Float16*)alloc(NLAY * convPackPerLayer * sizeof(_Float16));
  _Float16*  cpack2  = (_Float16*)alloc(NLAY * convPackPerLayer * sizeof(_Float16));
  _Float16*  qpack   = (_Float16*)alloc((size_t)CB_K * NT_N * FRAG_ELEMS * sizeof(_Float16));
  _Float16*  kpack   = (_Float16*)alloc((size_t)CB_K * NT_N * FRAG_ELEMS * sizeof(_Float16));
  _Float16*  vpack   = (_Float16*)alloc((size_t)CB_K * NT_N * FRAG_ELEMS * sizeof(_Float16));
  _Float16*  opack   = (_Float16*)alloc((size_t)CB_K * NT_N * FRAG_ELEMS * sizeof(_Float16));

  // ---- pre-swizzle all GEMM weights into WMMA B-fragment layout ----
  k_pack_frag<<<NLAY * 27 * CB_K * NT_N, 32, 0, stream>>>(conv1_w, cpack1);
  k_pack_frag<<<NLAY * 27 * CB_K * NT_N, 32, 0, stream>>>(conv2_w, cpack2);
  k_pack_frag<<<CB_K * NT_N, 32, 0, stream>>>(q_w, qpack);
  k_pack_frag<<<CB_K * NT_N, 32, 0, stream>>>(k_w, kpack);
  k_pack_frag<<<CB_K * NT_N, 32, 0, stream>>>(v_w, vpack);
  k_pack_frag<<<CB_K * NT_N, 32, 0, stream>>>(o_w, opack);

  dim3 gemmGrid(VT_M, NT_N);
  dim3 attnGrid(VT_M, NH);

  auto run_branch = [&](const float* pts, float* pf, _Float16* pfh) {
    k_fill_i32<<<1, 32, 0, stream>>>(cmin, 3, INT_MAX);
    k_fill_i32<<<(G3V + 1 + 255) / 256, 256, 0, stream>>>(lut, G3V + 1, 0);
    k_fill_i32<<<(NPTS + 255) / 256, 256, 0, stream>>>(cnts, NPTS, 0);
    k_coordmin<<<(NPTS + 255) / 256, 256, 0, stream>>>(pts, cmin);
    k_encode<<<NPTS, CF, 0, stream>>>(pts, enc_w, enc_b, enc_ln_g, enc_ln_b, feats);
    k_hashmark<<<(NPTS + 255) / 256, 256, 0, stream>>>(pts, cmin, hpt, lut);
    k_scan<<<1, 1024, 0, stream>>>(lut, vcoords, nvox);
    k_inv<<<(NPTS + 255) / 256, 256, 0, stream>>>(hpt, lut, inv, cnts);
    k_voxmean<<<NPTS, CF, 0, stream>>>(feats, inv, cnts, xv, valid, xvh);
    k_neighbors<<<NPTS, 32, 0, stream>>>(vcoords, valid, lut, nidx);
    for (int l = 0; l < NLAY; l++) {
      k_subm_conv<<<VT_M, 384, 0, stream>>>(xvh, nidx, cpack1 + (size_t)l * convPackPerLayer, tbuf);
      k_bnreduce<<<CF, 256, 0, stream>>>(tbuf, valid, bsum, bsq);
      k_bnapply<<<NPTS, CF, 0, stream>>>(tbuf, bsum, bsq, nvox, bn1_g + l * CF, bn1_b + l * CF,
                                         nullptr, nullptr, o1h, 1);
      k_subm_conv<<<VT_M, 384, 0, stream>>>(o1h, nidx, cpack2 + (size_t)l * convPackPerLayer, tbuf);
      k_bnreduce<<<CF, 256, 0, stream>>>(tbuf, valid, bsum, bsq);
      k_bnapply<<<NPTS, CF, 0, stream>>>(tbuf, bsum, bsq, nvox, bn2_g + l * CF, bn2_b + l * CF,
                                         xv, xv, xvh, 1);
    }
    k_gatherpts<<<NPTS, CF, 0, stream>>>(xv, inv, pf, pfh);
  };

  run_branch(src_pts, pf0, pf0h);
  run_branch(tgt_pts, pf1, pf1h);

  auto run_dir = [&](const float* pfS, const _Float16* pfSh, const _Float16* pfTh, int base) {
    k_gemm_proj<<<gemmGrid, 32, 0, stream>>>(pfSh, qpack, q_b, nullptr, nullptr, qhb, nullptr);
    k_gemm_proj<<<gemmGrid, 32, 0, stream>>>(pfTh, kpack, k_b, nullptr, nullptr, khb, nullptr);
    k_gemm_proj<<<gemmGrid, 32, 0, stream>>>(pfTh, vpack, v_b, nullptr, nullptr, nullptr, vT);
    k_attn_stats<<<attnGrid, 32, 0, stream>>>(qhb, khb, rowmax, rowsum);
    k_attn_out<<<attnGrid, 32, 0, stream>>>(qhb, khb, vT, rowmax, rowsum, ohb);
    k_gemm_proj<<<gemmGrid, 32, 0, stream>>>(ohb, opack, o_b, pfS, xo, nullptr, nullptr);
    k_layernorm<<<NPTS, CF, 0, stream>>>(xo, aln_g, aln_b);
    k_pool<<<CF, 256, 0, stream>>>(xo, fused, base);
  };

  run_dir(pf0, pf0h, pf1h, 0);        // cross_attn(source, target)
  run_dir(pf1, pf1h, pf0h, 2 * CF);   // cross_attn(target, source)

  k_mlp<<<(2 * CF + 63) / 64, 64, 0, stream>>>(fused, h1_w, h1_b, hb1, 4 * CF, 2 * CF, 1);
  k_mlp<<<(CF + 63) / 64, 64, 0, stream>>>(hb1, h2_w, h2_b, hb2, 2 * CF, CF, 1);
  k_mlp<<<1, 32, 0, stream>>>(hb2, h3_w, h3_b, (float*)d_out, CF, 9, 0);
}
